// NeRF_16295105921411
// MI455X (gfx1250) — matrix operand order repaired and checked by reading, not compiled, not run on hardware
//
#include <hip/hip_runtime.h>

// ---------------------------------------------------------------------------
// NeRF fused renderer for MI455X (gfx1250, wave32, WMMA).
//   inputs (setup_inputs order):
//     0 origins[2048*3] f32, 1 directions[2048*3] f32, 2 near f32[1], 3 far f32[1],
//     4 W0[63*256], 5 b0[256], 6 W1[256*256], 7 b1[256],
//     8 W2[256*256], 9 b2[256], 10 W3[256*4], 11 b3[4]      (all f32)
//   output: [2048*3] f32
// Strategy: weights pre-swizzled to fp16 WMMA-B fragment layout in d_ws,
// one workgroup per ray (64 samples), whole MLP fused through LDS,
// v_wmma_f32_16x16x32_f16 for every GEMM, composite in-workgroup.
// This revision: base-pointer + immediate-offset addressing so ds/global
// ops fold offsets into their 24-bit imm fields, and single-op ReLU.
// ---------------------------------------------------------------------------

typedef _Float16 v16h __attribute__((ext_vector_type(16)));
typedef _Float16 v8h  __attribute__((ext_vector_type(8)));
typedef float    v8f  __attribute__((ext_vector_type(8)));

#define NUM_SAMPLES 64
#define HIDDEN      256
#define NRAYS       2048

// ---------------- weight packing (fp32 row-major -> fp16 B-fragment) -------
// B-matrix 16-bit 32x16 VGPR layout (ISA 7.12.2): VGPR j, lanes 0-15 hold
// rows K = 2j,2j+1 (j<4), 8+2j,9+2j (j>=4) (cols = lane); lanes 16-31 hold those K + 8.
// Packed element ((kb*NT+nt)*32 + lane)*16 + e  <-  W[kb*32 + (lane>>4)*8 + (e&7) + (e>>3)*16][nt*16 + (lane&15)]
__global__ void pack_weights_kernel(const float* __restrict__ W,
                                    _Float16* __restrict__ Wp,
                                    int K0, int N0, int KB, int NT) {
    int idx = blockIdx.x * blockDim.x + threadIdx.x;
    int total = KB * NT * 32 * 16;
    if (idx >= total) return;
    int e    = idx & 15;
    int lane = (idx >> 4) & 31;
    int blk  = idx >> 9;
    int nt   = blk % NT;
    int kb   = blk / NT;
    int k    = kb * 32 + ((lane >> 4) << 3) + (e & 7) + ((e >> 3) << 4);
    int col  = nt * 16 + (lane & 15);
    float v  = (k < K0 && col < N0) ? W[k * N0 + col] : 0.0f;
    Wp[idx] = (_Float16)v;
}

// ---------------- fragment loaders ------------------------------------------
// A-matrix 16-bit 16x32 layout: lane L(<16) row M=L holds K {0..7, 16..23};
// lane L+16 row M=L holds K {8..15, 24..31}.  Two ds_load_b128 per lane.
__device__ __forceinline__ v16h ld_a_frag(const _Float16* p) {
    v8h lo = *(const v8h*)(p);       // K base + 0..7
    v8h hi = *(const v8h*)(p + 16);  // K base + 16..23
    return __builtin_shufflevector(lo, hi, 0,1,2,3,4,5,6,7,8,9,10,11,12,13,14,15);
}
// Packed B: 16 contiguous halves per lane -> two global_load_b128.
__device__ __forceinline__ v16h ld_b_frag(const _Float16* __restrict__ p) {
    v8h lo = *(const v8h*)(p);
    v8h hi = *(const v8h*)(p + 8);
    return __builtin_shufflevector(lo, hi, 0,1,2,3,4,5,6,7,8,9,10,11,12,13,14,15);
}

// ---------------- one hidden layer: [64 x KPAD] @ [KPAD x 256] + b, ReLU ----
// 8 waves: wave w -> M-tile (w&3), N-tiles [(w>>2)*8, +8).  sOut stride 256.
// All addresses are (per-lane base) + compile-time immediate offsets.
template <int KB, int ASTRIDE>
__device__ __forceinline__ void mlp_layer(const _Float16* sA, _Float16* sOut,
                                          const _Float16* __restrict__ Wp,
                                          const float* __restrict__ bias, int tid) {
    const int lane   = tid & 31;
    const int wave   = tid >> 5;
    const int mt     = wave & 3;
    const int ntBase = (wave >> 2) * 8;
    const int hiSel  = lane >> 4;           // 0 or 1
    const int bcol   = lane & 15;
    const int row    = mt * 16 + bcol;

    // lane-invariant bases; every access below uses a constant offset
    const _Float16* abase = sA + row * ASTRIDE + hiSel * 8;
    const _Float16* wbase = Wp + ((size_t)ntBase * 32 + lane) * 16;  // + bi*512
    const float*    bbase = bias + ntBase * 16 + bcol;               // + i*16
    _Float16*       obase = sOut + (mt * 16 + hiSel * 8) * 256 + ntBase * 16 + bcol;

    v16h a[KB];
#pragma unroll
    for (int kb = 0; kb < KB; ++kb) a[kb] = ld_a_frag(abase + kb * 32);

#pragma unroll
    for (int i = 0; i < 8; ++i) {
        const float bv = bbase[i * 16];
        v8f acc = {bv, bv, bv, bv, bv, bv, bv, bv};
#pragma unroll
        for (int kb = 0; kb < KB; ++kb) {
            // block index bi = kb*16 + (ntBase+i); ntBase folded into wbase
            v16h b = ld_b_frag(wbase + ((size_t)kb * 16 + i) * 512);
            acc = __builtin_amdgcn_wmma_f32_16x16x32_f16(
                false, a[kb], false, b, (short)0, acc, false, false);
        }
        // ReLU + fp16 store to LDS (C layout: VGPR r -> M = r + hiSel*8, N = bcol)
#pragma unroll
        for (int r = 0; r < 8; ++r) {
            float v = __builtin_fmaxf(acc[r], 0.0f);
            obase[r * 256 + i * 16] = (_Float16)v;
        }
    }
}

// ---------------- output layer: [64 x 256] @ [256 x 16(pad of 4)] + b3 ------
__device__ __forceinline__ void mlp_out_layer(const _Float16* sA, float* sF,
                                              const _Float16* __restrict__ W3p,
                                              const float* __restrict__ b3, int tid) {
    const int lane = tid & 31;
    const int wave = tid >> 5;
    if (wave >= 4) return;                  // wave-uniform: EXEC all-1s inside
    const int mt    = wave;
    const int hiSel = lane >> 4;
    const int bcol  = lane & 15;
    const int row   = mt * 16 + bcol;

    const float bv = (bcol < 4) ? b3[bcol] : 0.0f;
    v8f acc = {bv, bv, bv, bv, bv, bv, bv, bv};
    const _Float16* abase = sA + row * 256 + hiSel * 8;
    const _Float16* wbase = W3p + (size_t)lane * 16;
#pragma unroll
    for (int kb = 0; kb < 8; ++kb) {
        v16h a = ld_a_frag(abase + kb * 32);
        v16h b = ld_b_frag(wbase + (size_t)kb * 512);
        acc = __builtin_amdgcn_wmma_f32_16x16x32_f16(
            false, a, false, b, (short)0, acc, false, false);
    }
    float* fbase = sF + (mt * 16 + hiSel * 8) * 16 + bcol;
#pragma unroll
    for (int r = 0; r < 8; ++r)
        fbase[r * 16] = acc[r];
}

// ---------------- fused kernel: one workgroup == one ray --------------------
__global__ void __launch_bounds__(256, 2)
nerf_fused_kernel(const float* __restrict__ origins, const float* __restrict__ dirs,
                  const float* __restrict__ nearP,  const float* __restrict__ farP,
                  const _Float16* __restrict__ W0p, const float* __restrict__ b0,
                  const _Float16* __restrict__ W1p, const float* __restrict__ b1,
                  const _Float16* __restrict__ W2p, const float* __restrict__ b2,
                  const _Float16* __restrict__ W3p, const float* __restrict__ b3,
                  float* __restrict__ out) {
    // 64 KB static LDS: two ping-pong activation buffers; field aliases buf0.
    __shared__ __align__(16) unsigned char smem[65536];
    _Float16* sbuf0 = (_Float16*)smem;             // 64x256 fp16 (layer0 uses 64x64)
    _Float16* sbuf1 = (_Float16*)(smem + 32768);   // 64x256 fp16
    float*    sF    = (float*)smem;                // 64x16 fp32 (after layer2 done)

    const int tid = threadIdx.x;
    const int ray = blockIdx.x;

    const float nearv = *nearP;
    const float farv  = *farP;
    const float d     = (farv - nearv) * (1.0f / NUM_SAMPLES);   // linspace step == delta

    // ---- phase 1: positional encoding -> sbuf0 [64 x 64] fp16 --------------
    if (tid < NUM_SAMPLES) {
        const int s = tid;
        // faithful to source: mid = boundaries[s] + boundaries[s+1]/2
        const float mid = (nearv + s * d) + 0.5f * (nearv + (s + 1) * d);
        float loc[3];
#pragma unroll
        for (int c = 0; c < 3; ++c)
            loc[c] = origins[ray * 3 + c] + mid * dirs[ray * 3 + c];
        _Float16* f = sbuf0 + s * 64;
#pragma unroll
        for (int c = 0; c < 3; ++c) f[c] = (_Float16)loc[c];
        float freq = 1.0f;
#pragma unroll
        for (int l = 0; l < 10; ++l) {
#pragma unroll
            for (int c = 0; c < 3; ++c) {
                float xb = freq * loc[c];
                f[3 + 6 * l + c] = (_Float16)__sinf(xb);
                f[6 + 6 * l + c] = (_Float16)__cosf(xb);
            }
            freq *= 2.0f;
        }
        f[63] = (_Float16)0.0f;   // pad feature (W0 row 63 zeroed in packing)
    }
    __syncthreads();

    // ---- phases 2-4: MLP via WMMA ------------------------------------------
    mlp_layer<2, 64>(sbuf0, sbuf1, W0p, b0, tid);   // 63(64)->256
    __syncthreads();
    mlp_layer<8, 256>(sbuf1, sbuf0, W1p, b1, tid);  // 256->256
    __syncthreads();
    mlp_layer<8, 256>(sbuf0, sbuf1, W2p, b2, tid);  // 256->256
    __syncthreads();
    mlp_out_layer(sbuf1, sF, W3p, b3, tid);         // 256->4 (fp32 field)
    __syncthreads();

    // ---- phase 5: activation + alpha compositing ---------------------------
    if (tid < NUM_SAMPLES) {
        float* fr = sF + tid * 16;
        float sigma = __builtin_fmaxf(fr[3], 0.0f);
        float alpha = 1.0f - __expf(-sigma * d);
#pragma unroll
        for (int c = 0; c < 3; ++c)
            fr[c] = 1.0f / (1.0f + __expf(-fr[c]));   // sigmoid rgb
        fr[3] = alpha;
    }
    __syncthreads();
    if (tid == 0) {
        float accr = 0.f, accg = 0.f, accb = 0.f, T = 1.f;
#pragma unroll 4
        for (int s = 0; s < NUM_SAMPLES; ++s) {
            const float* fr = sF + s * 16;
            float a = fr[3];
            float w = T * a;
            accr += w * fr[0];
            accg += w * fr[1];
            accb += w * fr[2];
            T *= (1.0f - a);
        }
        out[ray * 3 + 0] = accr;
        out[ray * 3 + 1] = accg;
        out[ray * 3 + 2] = accb;
    }
}

// ---------------------------------------------------------------------------
extern "C" void kernel_launch(void* const* d_in, const int* in_sizes, int n_in,
                              void* d_out, int out_size, void* d_ws, size_t ws_size,
                              hipStream_t stream) {
    const float* origins = (const float*)d_in[0];
    const float* dirs    = (const float*)d_in[1];
    const float* nearP   = (const float*)d_in[2];
    const float* farP    = (const float*)d_in[3];
    const float* W0 = (const float*)d_in[4];  const float* b0 = (const float*)d_in[5];
    const float* W1 = (const float*)d_in[6];  const float* b1 = (const float*)d_in[7];
    const float* W2 = (const float*)d_in[8];  const float* b2 = (const float*)d_in[9];
    const float* W3 = (const float*)d_in[10]; const float* b3 = (const float*)d_in[11];
    float* out = (float*)d_out;

    // fp16 packed weights in workspace (total 303,104 bytes)
    unsigned char* ws = (unsigned char*)d_ws;
    _Float16* W0p = (_Float16*)(ws + 0);        // 32 KB
    _Float16* W1p = (_Float16*)(ws + 32768);    // 128 KB
    _Float16* W2p = (_Float16*)(ws + 163840);   // 128 KB
    _Float16* W3p = (_Float16*)(ws + 294912);   // 8 KB

    // repack every call (deterministic; ws is not preserved across calls)
    pack_weights_kernel<<<(2 * 16 * 512 + 255) / 256, 256, 0, stream>>>(W0, W0p, 63, 256, 2, 16);
    pack_weights_kernel<<<(8 * 16 * 512 + 255) / 256, 256, 0, stream>>>(W1, W1p, 256, 256, 8, 16);
    pack_weights_kernel<<<(8 * 16 * 512 + 255) / 256, 256, 0, stream>>>(W2, W2p, 256, 256, 8, 16);
    pack_weights_kernel<<<(8 * 1 * 512 + 255) / 256, 256, 0, stream>>>(W3, W3p, 256, 4, 8, 1);

    nerf_fused_kernel<<<NRAYS, 256, 0, stream>>>(origins, dirs, nearP, farP,
                                                 W0p, b0, W1p, b1, W2p, b2, W3p, b3,
                                                 out);
    (void)in_sizes; (void)n_in; (void)out_size; (void)ws_size;
}
